// CausalStateEncoder_75874892251457
// MI455X (gfx1250) — matrix operand order repaired and checked by reading, not compiled, not run on hardware
//
#include <hip/hip_runtime.h>
#include <hip/hip_bf16.h>

typedef __attribute__((ext_vector_type(16))) _Float16 v16h;
typedef __attribute__((ext_vector_type(8)))  float    v8f;

// ---------------- problem constants ----------------
constexpr int T  = 4096;
constexpr int Nv = 64;
constexpr int H  = 512;
constexpr int NN = 4096;        // N*N
constexpr int COMB = H + NN;    // 4608

constexpr int THREADS = 256;    // 8 wave32
constexpr int NWAVE   = 8;

// ---------------- workgroup roles ----------------
// role 0: GRU   (96 tiles, 18 chunks: 16 for W_hh(K=512) + 2 for W_ih(K=64))
// role 1: W1    (256 subtiles = 32 row-tiles x 8 K-slices of 576, 18 chunks each)
// role 2: W2    (32 tiles, 16 chunks)
// role 3: MU    (256 tiles, 16 chunks)
// role 4: LV    (256 tiles, 16 chunks)
constexpr int WG_GRU = 0;
constexpr int WG_W1  = 6;
constexpr int WG_W2  = 22;
constexpr int WG_MU  = 24;
constexpr int WG_LV  = 40;
constexpr int GRID   = 56;

// ---------------- fragment regions (units of 512-half frags) ----------------
constexpr long FB_GRU0 = 0;
constexpr long FB_W1_0 = 96L * 18;                 // 1728
constexpr long FB_W2_0 = FB_W1_0 + 256L * 18;      // 6336
constexpr long FB_MU_0 = FB_W2_0 + 32L * 16;       // 6848
constexpr long FB_LV_0 = FB_MU_0 + 256L * 16;      // 10944
constexpr long FB_END  = FB_LV_0 + 256L * 16;      // 15040 frags
constexpr long FR_TOTAL_HALFS = FB_END * 512;      // 7,700,480 halves (~15.4 MB)

// ---------------- control region in workspace (float indices) ----------------
constexpr int C_H0  = 0;      // h ping
constexpr int C_H1  = 512;    // h pong
constexpr int C_GI  = 1024;   // W_ih x + b_ih   (1536)
constexpr int C_GH  = 2560;   // W_hh h + b_hh   (1536)
constexpr int C_F1  = 4096;   // b1 + W1 comb    (512)
constexpr int C_F2  = 4608;   // b2 + W2 relu(f1)(512)
constexpr int C_CNT = 5120;   // barrier counter (unsigned)
constexpr int CTRL_FLOATS = 8192;

// ---------------- LDS layout ----------------
constexpr int LDS_FRAG_HALFS = 16 * 18 * 512;                  // 147456 (288 KB) max/WG
constexpr int SMEM_BYTES = LDS_FRAG_HALFS * 2 + COMB * 4;      // 313344 B <= 320 KB

// A-fragment index helpers (16-bit A 16x32, ISA 7.12.2)
__device__ __forceinline__ int frag_kl(int lane, int e) {
  int kh = (lane & 16) ? 8 : 0;
  return (e < 8) ? (kh + e) : (16 + kh + (e - 8));
}

// B-fragment: vector replicated into all 16 columns.
// lanes 0-15 hold K=0..7,16..23, lanes 16-31 hold K=8..15,24..31 of the 32-chunk.
__device__ __forceinline__ v16h make_b(const float* s, int kb, int lane) {
  const float* p = s + kb + ((lane & 16) ? 8 : 0);
  v16h b;
#pragma unroll
  for (int e = 0; e < 16; ++e) b[e] = (_Float16)p[(e < 8) ? e : (e + 8)];
  return b;
}

__device__ __forceinline__ v8f wmma16(v16h a, v16h b, v8f c) {
  return __builtin_amdgcn_wmma_f32_16x16x32_f16(false, a, false, b, (short)0, c,
                                                false, false);
}

__device__ __forceinline__ void gridbar(unsigned* cnt, unsigned target) {
  __threadfence();
  __syncthreads();
  if (threadIdx.x == 0) {
    __hip_atomic_fetch_add(cnt, 1u, __ATOMIC_RELEASE, __HIP_MEMORY_SCOPE_AGENT);
    while (__hip_atomic_load(cnt, __ATOMIC_ACQUIRE, __HIP_MEMORY_SCOPE_AGENT) < target)
      __builtin_amdgcn_s_sleep(2);
  }
  __syncthreads();
  __threadfence();
}

// ================= prep: zero state + barrier =================
__global__ void prep_state(float* ctrl) {
  int i = blockIdx.x * blockDim.x + threadIdx.x;
  if (i < 1024) ctrl[i] = 0.f;                      // h ping+pong = 0
  if (i == 0) *(unsigned*)(ctrl + C_CNT) = 0u;
}

// ================= prep: f32 weights -> f16 WMMA A-fragments =================
__global__ void prep_weights(const float* __restrict__ wih, const float* __restrict__ whh,
                             const float* __restrict__ w1,  const float* __restrict__ w2,
                             const float* __restrict__ muw, const float* __restrict__ lvw,
                             _Float16* __restrict__ frg) {
  long id = (long)blockIdx.x * blockDim.x + threadIdx.x;
  if (id >= FR_TOTAL_HALFS) return;
  long f   = id >> 9;              // fragment index
  int elem = (int)(id & 511);
  int lane = elem >> 4, e = elem & 15;
  int m  = lane & 15;
  int kl = frag_kl(lane, e);
  float v;
  if (f < FB_W1_0) {                                   // GRU: fused [W_hh | W_ih]
    int tile = (int)(f / 18), c = (int)(f % 18);
    int row = tile * 16 + m;
    v = (c < 16) ? whh[(long)row * 512 + c * 32 + kl]
                 : wih[(long)row * 64 + (c - 16) * 32 + kl];
  } else if (f < FB_W2_0) {                            // W1 subtiles
    long g = f - FB_W1_0;
    int S = (int)(g / 18), c = (int)(g % 18);
    int rt = S >> 3, s = S & 7;
    int row = rt * 16 + m, col = s * 576 + c * 32 + kl;
    v = w1[(long)row * COMB + col];
  } else if (f < FB_MU_0) {                            // W2
    long g = f - FB_W2_0;
    int tile = (int)(g / 16), c = (int)(g % 16);
    v = w2[(long)(tile * 16 + m) * 512 + c * 32 + kl];
  } else if (f < FB_LV_0) {                            // MU head
    long g = f - FB_MU_0;
    int tile = (int)(g / 16), c = (int)(g % 16);
    v = muw[(long)(tile * 16 + m) * 512 + c * 32 + kl];
  } else {                                             // LV head
    long g = f - FB_LV_0;
    int tile = (int)(g / 16), c = (int)(g % 16);
    v = lvw[(long)(tile * 16 + m) * 512 + c * 32 + kl];
  }
  frg[id] = (_Float16)v;
}

// ================= persistent scan kernel (3 barriers / step) =================
// Interval 1: W1 owners: gates(t) -> h(t), then W1 @ [h(t); z(t-1)] into f1
// Interval 2: W2 owners: f2 = W2 @ relu(f1)   ||  GRU owners: gi/gh(t+1) from h(t)
// Interval 3: head owners: mu/lv(t)           ||  W2 leader: f1 = b1 for t+1
__global__ void __launch_bounds__(THREADS)
csenc_main(const float* __restrict__ X,    const float* __restrict__ z_init,
           const float* __restrict__ b_ih, const float* __restrict__ b_hh,
           const float* __restrict__ b1v,  const float* __restrict__ b2v,
           const float* __restrict__ mu_b, const float* __restrict__ lv_b,
           float* __restrict__ out, float* ctrl, const _Float16* __restrict__ frg) {
  extern __shared__ char smem_raw[];
  _Float16* fr    = (_Float16*)smem_raw;
  float* scratch  = (float*)(smem_raw + (size_t)LDS_FRAG_HALFS * 2);

  const int wg = blockIdx.x, tid = threadIdx.x;
  const int lane = tid & 31, wave = tid >> 5;

  float* hA = ctrl + C_H0;
  float* hB = ctrl + C_H1;
  float* gi = ctrl + C_GI;
  float* gh = ctrl + C_GH;
  float* f1 = ctrl + C_F1;
  float* f2 = ctrl + C_F2;
  unsigned* cnt = (unsigned*)(ctrl + C_CNT);
  float* out_mu = out;
  float* out_lv = out + (long)T * NN;

  int role, local;
  long gbase;
  int nhalfs;
  if (wg < WG_W1)      { role = 0; local = wg;          gbase = (FB_GRU0 + (long)local * 16 * 18) * 512; nhalfs = 16 * 18 * 512; }
  else if (wg < WG_W2) { role = 1; local = wg - WG_W1;  gbase = (FB_W1_0 + (long)local * 16 * 18) * 512; nhalfs = 16 * 18 * 512; }
  else if (wg < WG_MU) { role = 2; local = wg - WG_W2;  gbase = (FB_W2_0 + (long)local * 16 * 16) * 512; nhalfs = 16 * 16 * 512; }
  else if (wg < WG_LV) { role = 3; local = wg - WG_MU;  gbase = (FB_MU_0 + (long)local * 16 * 16) * 512; nhalfs = 16 * 16 * 512; }
  else                 { role = 4; local = wg - WG_LV;  gbase = (FB_LV_0 + (long)local * 16 * 16) * 512; nhalfs = 16 * 16 * 512; }

  // one-time: pin this WG's weight fragments in LDS for all 4096 steps
  {
    const uint4* src = (const uint4*)(frg + gbase);
    uint4* dst = (uint4*)fr;
    int n16 = nhalfs >> 3;
    for (int i = tid; i < n16; i += THREADS) dst[i] = src[i];
  }
  __syncthreads();

  // GRU GEMV for step tt: gi/gh(tt) = W_ih x(tt) + b_ih, W_hh hsrc + b_hh
  auto do_gru = [&](int tt, const float* hsrc) {
    for (int i = tid; i < Nv; i += THREADS) scratch[i]      = X[(long)tt * Nv + i];
    for (int i = tid; i < H;  i += THREADS) scratch[Nv + i] = hsrc[i];
    __syncthreads();
    for (int lt = wave; lt < 16; lt += NWAVE) {
      int gt = local * 16 + lt;                 // global row-tile (0..95)
      v8f ah = {}; v8f ai = {};
#pragma unroll
      for (int c = 0; c < 16; ++c) {            // W_hh @ h
        v16h a = *(const v16h*)(fr + ((lt * 18 + c) << 9) + lane * 16);
        ah = wmma16(a, make_b(scratch, Nv + c * 32, lane), ah);
      }
#pragma unroll
      for (int c = 16; c < 18; ++c) {           // W_ih @ x
        v16h a = *(const v16h*)(fr + ((lt * 18 + c) << 9) + lane * 16);
        ai = wmma16(a, make_b(scratch, (c - 16) * 32, lane), ai);
      }
      if ((lane & 15) == 0) {                   // lanes 0,16 hold the result column
        int mb = (lane >> 4) * 8;
#pragma unroll
        for (int r = 0; r < 8; ++r) {
          int row = gt * 16 + mb + r;
          gi[row] = ai[r] + b_ih[row];
          gh[row] = ah[r] + b_hh[row];
        }
      }
    }
  };

  unsigned bar = 0;
  int p = 0;

  // ---- prologue: gi/gh(0) from h(-1)=0; f1 = b1 ----
  if (role == 0) {
    do_gru(0, hA);
  } else if (role == 2 && local == 0) {
    for (int i = tid; i < H; i += THREADS) f1[i] = b1v[i];
  }
  gridbar(cnt, ++bar * GRID);

  for (int t = 0; t < T; ++t) {
    const float* hcur = p ? hB : hA;   // h(t-1)
    float* hpub       = p ? hA : hB;   // h(t) destination

    // ---------------- I1: gates(t) + W1 @ [h(t) ; z(t-1)] ----------------
    if (role == 1) {
      for (int j = tid; j < H; j += THREADS) {    // GRU gate math (redundant per WG)
        float r = 1.f / (1.f + expf(-(gi[j] + gh[j])));
        float u = 1.f / (1.f + expf(-(gi[H + j] + gh[H + j])));
        float n = tanhf(gi[2 * H + j] + r * gh[2 * H + j]);
        float hv = (1.f - u) * n + u * hcur[j];
        scratch[j] = hv;
        if (wg == WG_W1) hpub[j] = hv;            // publish h(t)
      }
      const float* zsrc = (t == 0) ? z_init : (out_mu + (long)(t - 1) * NN);
      for (int i = tid; i < NN; i += THREADS) scratch[H + i] = zsrc[i];
      __syncthreads();
      for (int ls = wave; ls < 16; ls += NWAVE) {
        int S = local * 16 + ls;                  // subtile 0..255
        int rt = S >> 3, s = S & 7;
        v8f acc = {};
#pragma unroll
        for (int c = 0; c < 18; ++c) {
          v16h a = *(const v16h*)(fr + ((ls * 18 + c) << 9) + lane * 16);
          acc = wmma16(a, make_b(scratch, s * 576 + c * 32, lane), acc);
        }
        if ((lane & 15) == 0) {
          int mb = (lane >> 4) * 8;
#pragma unroll
          for (int r = 0; r < 8; ++r) atomicAdd(&f1[rt * 16 + mb + r], acc[r]);
        }
      }
    }
    gridbar(cnt, ++bar * GRID);

    // ---------------- I2: W2 @ relu(f1)  ||  GRU gi/gh(t+1) ----------------
    if (role == 2) {
      for (int i = tid; i < H; i += THREADS) scratch[i] = fmaxf(f1[i], 0.f);
      __syncthreads();
      for (int lt = wave; lt < 16; lt += NWAVE) {
        int gt = local * 16 + lt;                 // 0..31
        v8f acc = {};
#pragma unroll
        for (int c = 0; c < 16; ++c) {
          v16h a = *(const v16h*)(fr + ((lt * 16 + c) << 9) + lane * 16);
          acc = wmma16(a, make_b(scratch, c * 32, lane), acc);
        }
        if ((lane & 15) == 0) {
          int mb = (lane >> 4) * 8;
#pragma unroll
          for (int r = 0; r < 8; ++r) {
            int row = gt * 16 + mb + r;
            f2[row] = acc[r] + b2v[row];
          }
        }
      }
    } else if (role == 0 && (t + 1) < T) {
      do_gru(t + 1, hpub);                        // overlap with W2
    }
    gridbar(cnt, ++bar * GRID);

    // ---------------- I3: heads on relu(f2)  ||  f1 reset ----------------
    if (role >= 3) {
      for (int i = tid; i < H; i += THREADS) scratch[i] = fmaxf(f2[i], 0.f);
      __syncthreads();
      for (int lt = wave; lt < 16; lt += NWAVE) {
        int gt = local * 16 + lt;                 // 0..255
        v8f acc = {};
#pragma unroll
        for (int c = 0; c < 16; ++c) {
          v16h a = *(const v16h*)(fr + ((lt * 16 + c) << 9) + lane * 16);
          acc = wmma16(a, make_b(scratch, c * 32, lane), acc);
        }
        if ((lane & 15) == 0) {
          int mb = (lane >> 4) * 8;
#pragma unroll
          for (int r = 0; r < 8; ++r) {
            int row = gt * 16 + mb + r;
            if (role == 3) out_mu[(long)t * NN + row] = acc[r] + mu_b[row];
            else           out_lv[(long)t * NN + row] =
                fminf(fmaxf(acc[r] + lv_b[row], -6.f), 4.f);
          }
        }
      }
    } else if (role == 2 && local == 0) {
      for (int i = tid; i < H; i += THREADS) f1[i] = b1v[i];   // reset for t+1
    }
    gridbar(cnt, ++bar * GRID);   // publishes mu(t) as z for step t+1
    p ^= 1;
  }
}

// ================= host entry =================
extern "C" void kernel_launch(void* const* d_in, const int* in_sizes, int n_in,
                              void* d_out, int out_size, void* d_ws, size_t ws_size,
                              hipStream_t stream) {
  (void)in_sizes; (void)n_in; (void)out_size; (void)ws_size;
  const float* X    = (const float*)d_in[0];
  const float* zini = (const float*)d_in[1];
  const float* wih  = (const float*)d_in[2];
  const float* whh  = (const float*)d_in[3];
  const float* bih  = (const float*)d_in[4];
  const float* bhh  = (const float*)d_in[5];
  const float* w1   = (const float*)d_in[6];
  const float* b1   = (const float*)d_in[7];
  const float* w2   = (const float*)d_in[8];
  const float* b2   = (const float*)d_in[9];
  const float* muw  = (const float*)d_in[10];
  const float* mub  = (const float*)d_in[11];
  const float* lvw  = (const float*)d_in[12];
  const float* lvb  = (const float*)d_in[13];

  float* ctrl   = (float*)d_ws;
  _Float16* frg = (_Float16*)((char*)d_ws + (size_t)CTRL_FLOATS * 4);

  prep_state<<<4, THREADS, 0, stream>>>(ctrl);

  unsigned nblk = (unsigned)((FR_TOTAL_HALFS + THREADS - 1) / THREADS);
  prep_weights<<<nblk, THREADS, 0, stream>>>(wih, whh, w1, w2, muw, lvw, frg);

  (void)hipFuncSetAttribute((const void*)csenc_main,
                            hipFuncAttributeMaxDynamicSharedMemorySize, SMEM_BYTES);
  csenc_main<<<GRID, THREADS, SMEM_BYTES, stream>>>(
      X, zini, bih, bhh, b1, b2, mub, lvb, (float*)d_out, ctrl, frg);
}
